// KANLayer_83623013253734
// MI455X (gfx1250) — matrix operand order repaired and checked by reading, not compiled, not run on hardware
//
#include <hip/hip_runtime.h>
#include <hip/hip_bf16.h>

typedef _Float16 v8h  __attribute__((ext_vector_type(8)));
typedef _Float16 v16h __attribute__((ext_vector_type(16)));
typedef float    v8f  __attribute__((ext_vector_type(8)));

#define BATCH 2048
#define NIN   64
#define NOUT  64
#define NB    6              // cubic B-spline basis count (init_G + k = 6)
#define KDIM  448            // 64*6 spline features + 64 silu features
#define GPTS  10             // grid points per row

// ---------------------------------------------------------------------------
// Kernel 1: evaluate the 6 cubic B-spline bases + silu ONCE per (b, jin).
// Reference redundantly evaluates these 64x (once per n_out). We write:
//   Fh (f16): feature matrix (BATCH x KDIM) for the WMMA GEMM
//   Bf (f32): basis values (BATCH x NIN x NB) for the spl_reg reduction
// ---------------------------------------------------------------------------
__global__ void kan_basis(const float* __restrict__ x,
                          const float* __restrict__ grid,
                          _Float16* __restrict__ Fh,
                          float* __restrict__ Bf) {
    int idx = blockIdx.x * blockDim.x + threadIdx.x;   // 0 .. BATCH*NIN-1
    int b   = idx >> 6;
    int jin = idx & 63;
    float xv = x[idx];                                 // x is (batch, 64) row-major

    float tt[GPTS];
    const float* t = grid + jin * GPTS;                // all rows identical; row jin is exact
#pragma unroll
    for (int j = 0; j < GPTS; ++j) tt[j] = t[j];

    // Cox-de Boor, degree 0 -> 3
    float bas[9];
#pragma unroll
    for (int j = 0; j < 9; ++j)
        bas[j] = (xv >= tt[j] && xv < tt[j + 1]) ? 1.0f : 0.0f;
#pragma unroll
    for (int K = 1; K <= 3; ++K) {
#pragma unroll
        for (int j = 0; j < 9 - K; ++j) {
            float left  = (xv - tt[j])       / (tt[j + K]     - tt[j]);
            float right = (tt[j + K + 1] - xv) / (tt[j + K + 1] - tt[j + 1]);
            bas[j] = left * bas[j] + right * bas[j + 1];
        }
    }

    float silu = xv / (1.0f + __expf(-xv));

    _Float16* frow = Fh + (size_t)b * KDIM;
    float*    brow = Bf + (size_t)b * (NIN * NB) + jin * NB;
#pragma unroll
    for (int k = 0; k < NB; ++k) {
        frow[jin * NB + k] = (_Float16)bas[k];
        brow[k]            = bas[k];
    }
    frow[NIN * NB + jin] = (_Float16)silu;
}

// ---------------------------------------------------------------------------
// Kernel 2: fold c_spl / c_res / (1/n_in) into one f16 weight matrix
//   Wh (NOUT x KDIM):  Wh[o, jin*6+k] = c_spl[o*64+jin]*c_basis[o*64+jin, k]/64
//                      Wh[o, 384+jin] = c_res[o*64+jin]/64
// ---------------------------------------------------------------------------
__global__ void kan_weights(const float* __restrict__ c_basis,
                            const float* __restrict__ c_spl,
                            const float* __restrict__ c_res,
                            _Float16* __restrict__ Wh) {
    int idx = blockIdx.x * blockDim.x + threadIdx.x;
    if (idx >= NOUT * KDIM) return;
    int o  = idx / KDIM;
    int kc = idx - o * KDIM;
    float w;
    if (kc < NIN * NB) {
        int jin = kc / NB;
        int k   = kc - jin * NB;
        int i   = o * NIN + jin;
        w = c_spl[i] * c_basis[i * NB + k];
    } else {
        w = c_res[o * NIN + (kc - NIN * NB)];
    }
    Wh[idx] = (_Float16)(w * (1.0f / 64.0f));
}

// ---------------------------------------------------------------------------
// Kernel 3: y = Fh (2048x448) @ Wh^T (448x64) with v_wmma_f32_16x16x32_f16.
// 512 output tiles (128 M-tiles x 4 N-tiles), 8 waves/block, 64 blocks.
// Fragment loads follow the ISA 16-bit A 16x32 / B 32x16 wave32 layouts,
// each as two 16B global_load_b128 per lane.
// ---------------------------------------------------------------------------
__global__ void kan_gemm(const _Float16* __restrict__ Fh,
                         const _Float16* __restrict__ Wh,
                         float* __restrict__ y) {
    int wave  = threadIdx.x >> 5;
    int lane  = threadIdx.x & 31;
    int tile  = blockIdx.x * 8 + wave;    // 0..511
    int mtile = tile >> 2;                // 0..127
    int ntile = tile & 3;                 // 0..3
    int mn    = lane & 15;                // row for A, col for B/D
    int hi    = lane >> 4;                // lane-half

    const _Float16* arow = Fh + (size_t)(mtile * 16 + mn) * KDIM;
    const _Float16* brow = Wh + (size_t)(ntile * 16 + mn) * KDIM;

    v8f acc = {};
#pragma unroll
    for (int kb = 0; kb < KDIM; kb += 32) {
        // A 16x32: lanes 0-15 hold K {0..7, 16..23}, lanes 16-31 hold {8..15, 24..31}
        v8h a_lo = *(const v8h*)(arow + kb + hi * 8);
        v8h a_hi = *(const v8h*)(arow + kb + 16 + hi * 8);
        v16h a = __builtin_shufflevector(a_lo, a_hi,
                 0,1,2,3,4,5,6,7,8,9,10,11,12,13,14,15);
        // B 32x16: lanes 0-15 hold K {0..7, 16..23}, lanes 16-31 hold {8..15, 24..31} (col = lane&15)
        v8h b_lo = *(const v8h*)(brow + kb + hi * 8);
        v8h b_hi = *(const v8h*)(brow + kb + 16 + hi * 8);
        v16h bm = __builtin_shufflevector(b_lo, b_hi,
                 0,1,2,3,4,5,6,7,8,9,10,11,12,13,14,15);
        acc = __builtin_amdgcn_wmma_f32_16x16x32_f16(
                  false, a, false, bm, (short)0, acc, false, false);
    }

    // D layout: VGPR v <-> row M = v + 8*hi, col = lane&15
    float* yout = y + (size_t)(mtile * 16 + 8 * hi) * NOUT + ntile * 16 + mn;
#pragma unroll
    for (int v = 0; v < 8; ++v)
        yout[v * NOUT] = acc[v];
}

// ---------------------------------------------------------------------------
// Kernel 4: spl_reg[o, jin] = mean_b |dot6(c_basis[o*64+jin], B[b,jin,:])| / norm
// One block per jin; the (2048 x 6) basis slice lives in 48KB LDS (stride-6
// float addressing -> conflict-free across a wave32). Wave w handles 8 outputs,
// batch reduced lane-parallel then shfl_xor butterfly.
// ---------------------------------------------------------------------------
__global__ void kan_reg(const float* __restrict__ Bf,
                        const float* __restrict__ c_basis,
                        const float* __restrict__ grid,
                        float* __restrict__ out_reg) {
    __shared__ float Bsh[BATCH * NB];   // 48 KB of the 320 KB WGP LDS
    int jin = blockIdx.x;

    for (int e = threadIdx.x; e < BATCH * NB; e += blockDim.x) {
        int b = e / NB;
        int k = e - b * NB;
        Bsh[e] = Bf[(size_t)b * (NIN * NB) + jin * NB + k];
    }
    __syncthreads();

    int wave = threadIdx.x >> 5;
    int lane = threadIdx.x & 31;
#pragma unroll
    for (int t = 0; t < 8; ++t) {
        int o = wave + t * 8;
        int i = o * NIN + jin;
        float c0 = c_basis[i * NB + 0], c1 = c_basis[i * NB + 1];
        float c2 = c_basis[i * NB + 2], c3 = c_basis[i * NB + 3];
        float c4 = c_basis[i * NB + 4], c5 = c_basis[i * NB + 5];
        float acc = 0.0f;
        for (int b = lane; b < BATCH; b += 32) {
            const float* bb = Bsh + b * NB;
            float s = c0 * bb[0] + c1 * bb[1] + c2 * bb[2]
                    + c3 * bb[3] + c4 * bb[4] + c5 * bb[5];
            acc += fabsf(s);
        }
        acc += __shfl_xor(acc, 16);
        acc += __shfl_xor(acc, 8);
        acc += __shfl_xor(acc, 4);
        acc += __shfl_xor(acc, 2);
        acc += __shfl_xor(acc, 1);
        if (lane == 0) {
            float norm = grid[i * GPTS + 9] - grid[i * GPTS + 0] + 1e-5f;
            out_reg[i] = acc / (2048.0f * norm);
        }
    }
}

// ---------------------------------------------------------------------------
extern "C" void kernel_launch(void* const* d_in, const int* in_sizes, int n_in,
                              void* d_out, int out_size, void* d_ws, size_t ws_size,
                              hipStream_t stream) {
    const float* x       = (const float*)d_in[0];   // (2048, 64)
    const float* c_basis = (const float*)d_in[1];   // (4096, 6)
    const float* c_spl   = (const float*)d_in[2];   // (4096,)
    const float* c_res   = (const float*)d_in[3];   // (4096,)
    const float* grid    = (const float*)d_in[4];   // (4096, 10)
    float* out = (float*)d_out;                     // [y (2048*64) | spl_reg (64*64)]

    // Workspace carve-up (all 256B-aligned offsets)
    char* ws = (char*)d_ws;
    _Float16* Fh = (_Float16*)ws;                               // 2048*448*2   = 1,835,008 B
    _Float16* Wh = (_Float16*)(ws + 1835008);                   // 64*448*2     =    57,344 B
    float*    Bf = (float*)   (ws + 1835008 + 57344);           // 2048*64*6*4  = 3,145,728 B

    // 1. basis + silu features (one eval per (b, jin) instead of reference's 64x)
    kan_basis<<<(BATCH * NIN) / 256, 256, 0, stream>>>(x, grid, Fh, Bf);

    // 2. fused f16 weight matrix (c_spl*c_basis | c_res) / 64
    kan_weights<<<(NOUT * KDIM + 255) / 256, 256, 0, stream>>>(c_basis, c_spl, c_res, Wh);

    // 3. y = F @ W^T via WMMA (2048x448x64, fp32 accumulate)
    kan_gemm<<<64, 256, 0, stream>>>(Fh, Wh, out);

    // 4. spl_reg reduction
    kan_reg<<<NIN, 256, 0, stream>>>(Bf, c_basis, grid, out + BATCH * NOUT);
}
